// Seq2SeqWithAttention_22351009808386
// MI455X (gfx1250) — matrix operand order repaired and checked by reading, not compiled, not run on hardware
//
#include <hip/hip_runtime.h>

// ---------------- problem sizes ----------------
#define B_    16
#define S_    512
#define T_    128
#define E_    256
#define H_    512
#define H2_   1024
#define V_    32000
#define G4H_  (4 * H_)     // 2048
#define G4H2_ (4 * H2_)    // 4096
#define EH2_  (E_ + H2_)   // 1280
#define KDEC_ (E_ + 2 * H2_) // 2304 = [x | ctx | h]

typedef __bf16 bf16;
typedef __attribute__((ext_vector_type(16))) __bf16 v16bf;
typedef __attribute__((ext_vector_type(8)))  float  v8f;

struct alignas(16) Bf8   { bf16 e[8];  };
struct alignas(16) Bf16v { bf16 e[16]; };

__device__ __forceinline__ float sigm(float x) { return 1.0f / (1.0f + __expf(-x)); }

// =====================================================================
// Generic bf16 GEMM:  C[M][N] (f32) = A[M][K](bf16,lda) * W[N][K]^T (bf16,ldw)
//                               (+ bias[N]) (+ old C if accumulate)
// blockDim = 32 (one wave).  grid = (N/64, M/16).  K % 32 == 0.
// C/D layout per CDNA5 ISA 7.12.2 (wave32); A/B use contiguous k = 16*half + s.
// =====================================================================
__global__ void gemm_bf16(const bf16* __restrict__ A, int lda,
                          const bf16* __restrict__ W, int ldw,
                          const float* __restrict__ bias,
                          float* __restrict__ C, long ldc,
                          int K, int accumulate)
{
    const int lane = threadIdx.x & 31;
    const int lr   = lane & 15;   // row (A) / column (B,C) within tile
    const int half = lane >> 4;   // half-wave selector
    const int m0   = blockIdx.y << 4;
    const int n0   = blockIdx.x << 6;

    const bf16* Arow = A + (long)(m0 + lr) * lda;
    const bf16* W0 = W + (long)(n0 +  0 + lr) * ldw;
    const bf16* W1 = W + (long)(n0 + 16 + lr) * ldw;
    const bf16* W2 = W + (long)(n0 + 32 + lr) * ldw;
    const bf16* W3 = W + (long)(n0 + 48 + lr) * ldw;

    v8f acc0 = {}, acc1 = {}, acc2 = {}, acc3 = {};

    for (int k0 = 0; k0 < K; k0 += 32) {
        // A 16x32 bf16: lanes 0-15 hold K 0..15; lanes 16-31 hold K 16..31
        union { v16bf v; Bf8 s[2]; } a;
        a.s[0] = *(const Bf8*)(Arow + k0 +     half * 16);
        a.s[1] = *(const Bf8*)(Arow + k0 + 8 + half * 16);
        // B 32x16 bf16: lane = column; lanes 0-15 hold K 0..15, lanes 16-31 hold K 16..31
        const int kw = k0 + half * 16;
        union { v16bf v; Bf16v s; } b0, b1, b2, b3;
        b0.s = *(const Bf16v*)(W0 + kw);
        b1.s = *(const Bf16v*)(W1 + kw);
        b2.s = *(const Bf16v*)(W2 + kw);
        b3.s = *(const Bf16v*)(W3 + kw);

        acc0 = __builtin_amdgcn_wmma_f32_16x16x32_bf16(false, a.v, false, b0.v, (short)0, acc0, false, false);
        acc1 = __builtin_amdgcn_wmma_f32_16x16x32_bf16(false, a.v, false, b1.v, (short)0, acc1, false, false);
        acc2 = __builtin_amdgcn_wmma_f32_16x16x32_bf16(false, a.v, false, b2.v, (short)0, acc2, false, false);
        acc3 = __builtin_amdgcn_wmma_f32_16x16x32_bf16(false, a.v, false, b3.v, (short)0, acc3, false, false);
    }

    // C/D 16x16 f32: VGPR r -> row r (lanes 0-15) / row r+8 (lanes 16-31); col = lane%16
    for (int r = 0; r < 8; ++r) {
        const long ro = (long)(m0 + (half << 3) + r) * ldc;
        const int c0 = n0 + lr, c1 = c0 + 16, c2 = c0 + 32, c3 = c0 + 48;
        float v0 = acc0[r], v1 = acc1[r], v2 = acc2[r], v3 = acc3[r];
        if (bias)       { v0 += bias[c0];  v1 += bias[c1];  v2 += bias[c2];  v3 += bias[c3]; }
        if (accumulate) { v0 += C[ro+c0];  v1 += C[ro+c1];  v2 += C[ro+c2];  v3 += C[ro+c3]; }
        C[ro+c0] = v0; C[ro+c1] = v1; C[ro+c2] = v2; C[ro+c3] = v3;
    }
}

// ---------------- utility / elementwise kernels ----------------
__global__ void cast_f32_bf16(const float* __restrict__ src, bf16* __restrict__ dst, int n) {
    int i = blockIdx.x * 256 + threadIdx.x;
    if (i < n) dst[i] = (bf16)src[i];
}

__global__ void add_bias2(const float* __restrict__ a, const float* __restrict__ b,
                          float* __restrict__ out, int n) {
    int i = blockIdx.x * 256 + threadIdx.x;
    if (i < n) out[i] = a[i] + b[i];
}

__global__ void zero_f32(float* p, int n)  { int i = blockIdx.x * 256 + threadIdx.x; if (i < n) p[i] = 0.0f; }
__global__ void zero_bf(bf16* p, int n)    { int i = blockIdx.x * 256 + threadIdx.x; if (i < n) p[i] = (bf16)0.0f; }

__global__ void zero_out_t0(float* __restrict__ out) { // out[b][0][v] = 0
    int i = blockIdx.x * 256 + threadIdx.x;            // B_*V_
    if (i < B_ * V_) {
        int b = i / V_, v = i % V_;
        out[(long)b * T_ * V_ + v] = 0.0f;
    }
}

// Pack decoder weights: Wdec[n][0:1280] = Wih_d[n][:], [1280:2304] = Whh_d[n][:]
__global__ void pack_wdec(const float* __restrict__ Wih, const float* __restrict__ Whh,
                          bf16* __restrict__ dst) {
    int i = blockIdx.x * 256 + threadIdx.x;
    if (i >= G4H2_ * KDEC_) return;
    int n = i / KDEC_, k = i % KDEC_;
    float v = (k < EH2_) ? Wih[(long)n * EH2_ + k] : Whh[(long)n * H2_ + (k - EH2_)];
    dst[i] = (bf16)v;
}

// Embedding gathers:  X[s*B + b][E] = emb[src[b][s]]
__global__ void gather_src(const int* __restrict__ src, const float* __restrict__ emb,
                           bf16* __restrict__ X) {
    int i = blockIdx.x * 256 + threadIdx.x;       // S*B*E
    if (i >= S_ * B_ * E_) return;
    int e = i % E_; int sb = i / E_; int b = sb % B_; int s = sb / B_;
    int tok = src[b * S_ + s];
    X[i] = (bf16)emb[(long)tok * E_ + e];
}
__global__ void gather_tgt(const int* __restrict__ tgt, const float* __restrict__ emb,
                           bf16* __restrict__ X) {
    int i = blockIdx.x * 256 + threadIdx.x;       // (T-1)*B*E
    if (i >= (T_ - 1) * B_ * E_) return;
    int e = i % E_; int tb = i / E_; int b = tb % B_; int t = tb / B_;
    int tok = tgt[b * T_ + t];
    X[i] = (bf16)emb[(long)tok * E_ + e];
}

// Encoder LSTM gate activation, both directions fused. step index = st.
// Gxf_s/Gxb_s point at this step's [B][4H] pre-activation (x-part + bias + h-part).
__global__ void lstm_act_enc(const float* __restrict__ Gxf_s, const float* __restrict__ Gxb_s,
                             float* __restrict__ cf, float* __restrict__ cb,
                             bf16* __restrict__ hf, bf16* __restrict__ hb,
                             float* __restrict__ enc_out, bf16* __restrict__ enc_out_bf,
                             int st)
{
    int i = blockIdx.x * 256 + threadIdx.x;       // 2*B*H
    if (i >= 2 * B_ * H_) return;
    int dir = i / (B_ * H_);
    int j = i % (B_ * H_);
    int b = j / H_, hI = j % H_;
    const float* Gx = dir ? Gxb_s : Gxf_s;
    float zi = Gx[b * G4H_ + hI];
    float zf = Gx[b * G4H_ + H_ + hI];
    float zg = Gx[b * G4H_ + 2 * H_ + hI];
    float zo = Gx[b * G4H_ + 3 * H_ + hI];
    float* c = dir ? cb : cf;
    float cv = sigm(zf) * c[j] + sigm(zi) * tanhf(zg);
    c[j] = cv;
    float hv = sigm(zo) * tanhf(cv);
    (dir ? hb : hf)[j] = (bf16)hv;
    int spos = dir ? (S_ - 1 - st) : st;
    long off = ((long)b * S_ + spos) * H2_ + dir * H_ + hI;
    enc_out[off]    = hv;
    enc_out_bf[off] = (bf16)hv;
}

// Decoder initial state: h = [hTf | hTb] (bf16), c = [cTf | cTb] (f32)
__global__ void concat_state(const bf16* __restrict__ hf, const bf16* __restrict__ hb,
                             const float* __restrict__ cf, const float* __restrict__ cb,
                             bf16* __restrict__ hdec, float* __restrict__ cdec)
{
    int i = blockIdx.x * 256 + threadIdx.x;       // B*H2
    if (i >= B_ * H2_) return;
    int b = i / H2_, j = i % H2_;
    if (j < H_) { hdec[i] = hf[b * H_ + j];        cdec[i] = cf[b * H_ + j]; }
    else        { hdec[i] = hb[b * H_ + (j - H_)]; cdec[i] = cb[b * H_ + (j - H_)]; }
}

// energy[b][s] = sum_j tanh(enc_proj[b][s][j] + q[b][j]); one wave per row
__global__ void attn_energy(const float* __restrict__ enc_proj, const float* __restrict__ q,
                            float* __restrict__ energy)
{
    int wave = threadIdx.x >> 5, lane = threadIdx.x & 31;
    int row = blockIdx.x * 8 + wave;              // row = b*S + s, rows = B*S
    if (row >= B_ * S_) return;
    int b = row / S_;
    const float* ep = enc_proj + (long)row * H2_;
    const float* qb = q + b * H2_;
    float sum = 0.0f;
    for (int j = lane; j < H2_; j += 32) sum += tanhf(ep[j] + qb[j]);
    for (int m = 16; m >= 1; m >>= 1) sum += __shfl_xor(sum, m, 32);
    if (lane == 0) energy[row] = sum;
}

// softmax over S per batch; one block per b
__global__ void attn_softmax(const float* __restrict__ energy, float* __restrict__ w)
{
    __shared__ float red[256];
    int b = blockIdx.x, t = threadIdx.x;
    float m = -1e30f;
    for (int s = t; s < S_; s += 256) m = fmaxf(m, energy[b * S_ + s]);
    red[t] = m; __syncthreads();
    for (int st = 128; st > 0; st >>= 1) { if (t < st) red[t] = fmaxf(red[t], red[t + st]); __syncthreads(); }
    m = red[0]; __syncthreads();
    float sum = 0.0f;
    for (int s = t; s < S_; s += 256) { float e = __expf(energy[b * S_ + s] - m); w[b * S_ + s] = e; sum += e; }
    red[t] = sum; __syncthreads();
    for (int st = 128; st > 0; st >>= 1) { if (t < st) red[t] += red[t + st]; __syncthreads(); }
    float inv = 1.0f / red[0];
    __syncthreads();
    for (int s = t; s < S_; s += 256) w[b * S_ + s] *= inv;
}

// ctx[b][j] = sum_s w[b][s] * enc_out[b][s][j]
__global__ void attn_ctx(const float* __restrict__ w, const float* __restrict__ enc_out,
                         float* __restrict__ ctx)
{
    int b = blockIdx.x / (H2_ / 256);
    int j = (blockIdx.x % (H2_ / 256)) * 256 + threadIdx.x;
    const float* eo = enc_out + (long)b * S_ * H2_ + j;
    const float* wb = w + b * S_;
    float sum = 0.0f;
    for (int s = 0; s < S_; ++s) sum += wb[s] * eo[(long)s * H2_];
    ctx[b * H2_ + j] = sum;
}

// xcat[b][2304] = [ x_t(bf16,E) | ctx(f32->bf16,H2) | h_{t-1}(bf16,H2) ]
__global__ void build_xcat(const bf16* __restrict__ xdec_t, const float* __restrict__ ctx,
                           const bf16* __restrict__ hdec, bf16* __restrict__ xcat)
{
    int i = blockIdx.x * 256 + threadIdx.x;       // B*KDEC
    if (i >= B_ * KDEC_) return;
    int b = i / KDEC_, k = i % KDEC_;
    bf16 v;
    if (k < E_)         v = xdec_t[b * E_ + k];
    else if (k < EH2_)  v = (bf16)ctx[b * H2_ + (k - E_)];
    else                v = hdec[b * H2_ + (k - EH2_)];
    xcat[i] = v;
}

// decoder LSTM gate activation
__global__ void lstm_act_dec(const float* __restrict__ z, float* __restrict__ cdec,
                             bf16* __restrict__ hdec)
{
    int i = blockIdx.x * 256 + threadIdx.x;       // B*H2
    if (i >= B_ * H2_) return;
    int b = i / H2_, j = i % H2_;
    const float* zb = z + (long)b * G4H2_;
    float zi = zb[j], zf = zb[H2_ + j], zg = zb[2 * H2_ + j], zo = zb[3 * H2_ + j];
    float cv = sigm(zf) * cdec[i] + sigm(zi) * tanhf(zg);
    cdec[i] = cv;
    hdec[i] = (bf16)(sigm(zo) * tanhf(cv));
}

// ---------------- host side ----------------
static inline void* bump(char*& p, size_t bytes) {
    void* r = (void*)p;
    p += (bytes + 255) & ~(size_t)255;
    return r;
}

extern "C" void kernel_launch(void* const* d_in, const int* in_sizes, int n_in,
                              void* d_out, int out_size, void* d_ws, size_t ws_size,
                              hipStream_t stream)
{
    (void)in_sizes; (void)n_in; (void)out_size; (void)ws_size;
    const int*   src   = (const int*)  d_in[0];
    const int*   tgt   = (const int*)  d_in[1];
    const float* emb   = (const float*)d_in[2];
    const float* Wih_f = (const float*)d_in[3];
    const float* Whh_f = (const float*)d_in[4];
    const float* bih_f = (const float*)d_in[5];
    const float* bhh_f = (const float*)d_in[6];
    const float* Wih_b = (const float*)d_in[7];
    const float* Whh_b = (const float*)d_in[8];
    const float* bih_b = (const float*)d_in[9];
    const float* bhh_b = (const float*)d_in[10];
    const float* Wih_d = (const float*)d_in[11];
    const float* Whh_d = (const float*)d_in[12];
    const float* bih_d = (const float*)d_in[13];
    const float* bhh_d = (const float*)d_in[14];
    const float* Wa    = (const float*)d_in[15];
    const float* ba    = (const float*)d_in[16];
    const float* Wo    = (const float*)d_in[17];
    const float* bo    = (const float*)d_in[18];
    float* out = (float*)d_out;

    // ---- workspace layout ----
    char* p = (char*)d_ws;
    bf16*  Xbf      = (bf16*) bump(p, (size_t)S_ * B_ * E_ * 2);
    bf16*  Xdec     = (bf16*) bump(p, (size_t)(T_ - 1) * B_ * E_ * 2);
    float* Gx_f     = (float*)bump(p, (size_t)S_ * B_ * G4H_ * 4);
    float* Gx_b     = (float*)bump(p, (size_t)S_ * B_ * G4H_ * 4);
    float* enc_out  = (float*)bump(p, (size_t)B_ * S_ * H2_ * 4);
    bf16*  enc_obf  = (bf16*) bump(p, (size_t)B_ * S_ * H2_ * 2);
    float* enc_proj = (float*)bump(p, (size_t)B_ * S_ * H2_ * 4);
    bf16*  Wihf_bf  = (bf16*) bump(p, (size_t)G4H_ * E_ * 2);
    bf16*  Whhf_bf  = (bf16*) bump(p, (size_t)G4H_ * H_ * 2);
    bf16*  Wihb_bf  = (bf16*) bump(p, (size_t)G4H_ * E_ * 2);
    bf16*  Whhb_bf  = (bf16*) bump(p, (size_t)G4H_ * H_ * 2);
    bf16*  Wa_bf    = (bf16*) bump(p, (size_t)H2_ * 2 * H2_ * 2);
    bf16*  Wdec_bf  = (bf16*) bump(p, (size_t)G4H2_ * KDEC_ * 2);
    bf16*  Wo_bf    = (bf16*) bump(p, (size_t)V_ * H2_ * 2);
    float* bgate_f  = (float*)bump(p, (size_t)G4H_ * 4);
    float* bgate_b  = (float*)bump(p, (size_t)G4H_ * 4);
    float* bdec     = (float*)bump(p, (size_t)G4H2_ * 4);
    float* c_f      = (float*)bump(p, (size_t)B_ * H_ * 4);
    float* c_b      = (float*)bump(p, (size_t)B_ * H_ * 4);
    bf16*  h_f      = (bf16*) bump(p, (size_t)B_ * H_ * 2);
    bf16*  h_b      = (bf16*) bump(p, (size_t)B_ * H_ * 2);
    bf16*  hdec     = (bf16*) bump(p, (size_t)B_ * H2_ * 2);
    float* cdec     = (float*)bump(p, (size_t)B_ * H2_ * 4);
    float* qbuf     = (float*)bump(p, (size_t)B_ * H2_ * 4);
    float* energy   = (float*)bump(p, (size_t)B_ * S_ * 4);
    float* wgt      = (float*)bump(p, (size_t)B_ * S_ * 4);
    float* ctx      = (float*)bump(p, (size_t)B_ * H2_ * 4);
    bf16*  xcat     = (bf16*) bump(p, (size_t)B_ * KDEC_ * 2);
    float* zbuf     = (float*)bump(p, (size_t)B_ * G4H2_ * 4);

    const dim3 blk256(256), blk32(32);
    auto g1 = [](int n) { return dim3((n + 255) / 256); };

    // ---- one-time prep: weight casts, bias combines, gathers, zeros ----
    cast_f32_bf16<<<g1(G4H_ * E_),  blk256, 0, stream>>>(Wih_f, Wihf_bf, G4H_ * E_);
    cast_f32_bf16<<<g1(G4H_ * H_),  blk256, 0, stream>>>(Whh_f, Whhf_bf, G4H_ * H_);
    cast_f32_bf16<<<g1(G4H_ * E_),  blk256, 0, stream>>>(Wih_b, Wihb_bf, G4H_ * E_);
    cast_f32_bf16<<<g1(G4H_ * H_),  blk256, 0, stream>>>(Whh_b, Whhb_bf, G4H_ * H_);
    cast_f32_bf16<<<g1(H2_ * 2 * H2_), blk256, 0, stream>>>(Wa, Wa_bf, H2_ * 2 * H2_);
    cast_f32_bf16<<<g1(V_ * H2_),   blk256, 0, stream>>>(Wo, Wo_bf, V_ * H2_);
    pack_wdec<<<g1(G4H2_ * KDEC_), blk256, 0, stream>>>(Wih_d, Whh_d, Wdec_bf);
    add_bias2<<<g1(G4H_),  blk256, 0, stream>>>(bih_f, bhh_f, bgate_f, G4H_);
    add_bias2<<<g1(G4H_),  blk256, 0, stream>>>(bih_b, bhh_b, bgate_b, G4H_);
    add_bias2<<<g1(G4H2_), blk256, 0, stream>>>(bih_d, bhh_d, bdec, G4H2_);
    gather_src<<<g1(S_ * B_ * E_), blk256, 0, stream>>>(src, emb, Xbf);
    gather_tgt<<<g1((T_ - 1) * B_ * E_), blk256, 0, stream>>>(tgt, emb, Xdec);
    zero_f32<<<g1(B_ * H_), blk256, 0, stream>>>(c_f, B_ * H_);
    zero_f32<<<g1(B_ * H_), blk256, 0, stream>>>(c_b, B_ * H_);
    zero_bf <<<g1(B_ * H_), blk256, 0, stream>>>(h_f, B_ * H_);
    zero_bf <<<g1(B_ * H_), blk256, 0, stream>>>(h_b, B_ * H_);
    zero_out_t0<<<g1(B_ * V_), blk256, 0, stream>>>(out);

    // ---- encoder input projections (one big WMMA GEMM per direction) ----
    // Gx[s*B+b][4H] = Xbf @ Wih^T + (bih + bhh)
    gemm_bf16<<<dim3(G4H_ / 64, (S_ * B_) / 16), blk32, 0, stream>>>(
        Xbf, E_, Wihf_bf, E_, bgate_f, Gx_f, G4H_, E_, 0);
    gemm_bf16<<<dim3(G4H_ / 64, (S_ * B_) / 16), blk32, 0, stream>>>(
        Xbf, E_, Wihb_bf, E_, bgate_b, Gx_b, G4H_, E_, 0);

    // ---- bidirectional encoder scan (sequential over S) ----
    for (int st = 0; st < S_; ++st) {
        float* gf = Gx_f + (long)st * B_ * G4H_;
        float* gb = Gx_b + (long)(S_ - 1 - st) * B_ * G4H_;
        // z += h @ Whh^T (accumulate into precomputed x-part)
        gemm_bf16<<<dim3(G4H_ / 64, 1), blk32, 0, stream>>>(
            h_f, H_, Whhf_bf, H_, nullptr, gf, G4H_, H_, 1);
        gemm_bf16<<<dim3(G4H_ / 64, 1), blk32, 0, stream>>>(
            h_b, H_, Whhb_bf, H_, nullptr, gb, G4H_, H_, 1);
        lstm_act_enc<<<g1(2 * B_ * H_), blk256, 0, stream>>>(
            gf, gb, c_f, c_b, h_f, h_b, enc_out, enc_obf, st);
    }
    concat_state<<<g1(B_ * H2_), blk256, 0, stream>>>(h_f, h_b, c_f, c_b, hdec, cdec);

    // ---- attention key precompute: enc_proj = enc_out @ Wa_e^T + ba ----
    gemm_bf16<<<dim3(H2_ / 64, (B_ * S_) / 16), blk32, 0, stream>>>(
        enc_obf, H2_, Wa_bf + H2_, 2 * H2_, ba, enc_proj, H2_, H2_, 0);

    // ---- decoder loop ----
    for (int t = 0; t < T_ - 1; ++t) {
        // q = h @ Wa_h^T
        gemm_bf16<<<dim3(H2_ / 64, 1), blk32, 0, stream>>>(
            hdec, H2_, Wa_bf, 2 * H2_, nullptr, qbuf, H2_, H2_, 0);
        attn_energy<<<dim3((B_ * S_) / 8), blk256, 0, stream>>>(enc_proj, qbuf, energy);
        attn_softmax<<<dim3(B_), blk256, 0, stream>>>(energy, wgt);
        attn_ctx<<<dim3(B_ * (H2_ / 256)), blk256, 0, stream>>>(wgt, enc_out, ctx);
        build_xcat<<<g1(B_ * KDEC_), blk256, 0, stream>>>(
            Xdec + (long)t * B_ * E_, ctx, hdec, xcat);
        // z = [x|ctx|h] @ [Wih_d;Whh_d]^T + (bih_d + bhh_d)
        gemm_bf16<<<dim3(G4H2_ / 64, 1), blk32, 0, stream>>>(
            xcat, KDEC_, Wdec_bf, KDEC_, bdec, zbuf, G4H2_, KDEC_, 0);
        lstm_act_dec<<<g1(B_ * H2_), blk256, 0, stream>>>(zbuf, cdec, hdec);
        // pred = h @ Wo^T + bo  ->  out[:, t+1, :]
        gemm_bf16<<<dim3(V_ / 64, 1), blk32, 0, stream>>>(
            hdec, H2_, Wo_bf, H2_, bo, out + (long)(t + 1) * V_, (long)T_ * V_, H2_, 0);
    }
}
